// MLA_20401094656190
// MI455X (gfx1250) — compile-verified, matrix operand order checked by reading, not run on hardware
//
#include <hip/hip_runtime.h>
#include <stdint.h>

// ---------------- problem constants ----------------
#define B_      2
#define S_      2048
#define DIM_    2048
#define H_      16
#define KV_RANK_ 512
#define D_NOPE_ 128
#define D_ROPE_ 64
#define D_V_    128
#define QK_HEAD_ 192
#define M_TOT   (B_ * S_)      // 4096 token rows
#define NQ_     (H_ * QK_HEAD_) // 3072

typedef __attribute__((ext_vector_type(16))) __bf16 v16bf;
typedef __attribute__((ext_vector_type(8)))  __bf16 v8bf;
typedef __attribute__((ext_vector_type(8)))  float  v8f;

// ---------------- bf16 helpers ----------------
static __device__ __forceinline__ unsigned short f2bf(float f) {
    union { float f; unsigned int u; } c; c.f = f;
    unsigned int u = c.u;
    unsigned int r = 0x7FFFu + ((u >> 16) & 1u);   // round-to-nearest-even
    return (unsigned short)((u + r) >> 16);
}
static __device__ __forceinline__ float bf2f(unsigned short h) {
    union { unsigned int u; float f; } c; c.u = ((unsigned int)h) << 16;
    return c.f;
}

// Load one WMMA 16x32 bf16 fragment chunk-pair for this lane.
// base must point at row_start + kBase + (lane>=16 ? 8 : 0).
static __device__ __forceinline__ v16bf load_frag(const unsigned short* base) {
    v8bf lo = *(const v8bf*)(base);
    v8bf hi = *(const v8bf*)(base + 16);
    return __builtin_shufflevector(lo, hi, 0,1,2,3,4,5,6,7,8,9,10,11,12,13,14,15);
}

static __device__ __forceinline__ v8f wmma_bf16(v16bf a, v16bf b, v8f c) {
    return __builtin_amdgcn_wmma_f32_16x16x32_bf16(
        /*neg_a=*/false, a, /*neg_b=*/false, b,
        /*c_mod=*/(short)0, c, /*reuse_a=*/false, /*reuse_b=*/false);
}

static __device__ __forceinline__ void store_elem(float* p, float v)          { *p = v; }
static __device__ __forceinline__ void store_elem(unsigned short* p, float v) { *p = f2bf(v); }

// ---------------- generic tiled WMMA GEMM: C[M,N] = A[M,K] * W[N,K]^T ----------------
// 256 threads = 8 waves in a 2(M) x 4(N) grid; each wave computes 32x64.
template <typename OutT>
__global__ __launch_bounds__(256) void gemm_wmma_kernel(
    const unsigned short* __restrict__ A, long lda, long aBatch,
    const unsigned short* __restrict__ W, long ldb, long bBatch,
    OutT* __restrict__ C, long ldc, long cBatch,
    int M, int N, int K)
{
    long z = blockIdx.z;
    A += z * aBatch; W += z * bBatch; C += z * cBatch;

    int wave = threadIdx.x >> 5;
    int lane = threadIdx.x & 31;
    int lh   = lane >> 4;          // which 16-lane half
    int l16  = lane & 15;
    int koff = lh * 8;

    int mw = wave >> 2, nw = wave & 3;
    int mBase = blockIdx.y * 64  + mw * 32;
    int nBase = blockIdx.x * 256 + nw * 64;
    if (mBase >= M) return;        // wave-uniform

    bool nOk[4];
    #pragma unroll
    for (int j = 0; j < 4; j++) nOk[j] = (nBase + j * 16 + 16) <= N;

    v8f acc[2][4];
    #pragma unroll
    for (int i = 0; i < 2; i++)
        #pragma unroll
        for (int j = 0; j < 4; j++)
            acc[i][j] = (v8f){0.f,0.f,0.f,0.f,0.f,0.f,0.f,0.f};

    const unsigned short* aRow0 = A + (long)(mBase      + l16) * lda + koff;
    const unsigned short* aRow1 = A + (long)(mBase + 16 + l16) * lda + koff;
    const unsigned short* bRow[4];
    #pragma unroll
    for (int j = 0; j < 4; j++)
        bRow[j] = W + (long)((nOk[j] ? (nBase + j * 16) : 0) + l16) * ldb + koff;

    for (int k = 0; k < K; k += 32) {
        v16bf a0 = load_frag(aRow0 + k);
        v16bf a1 = load_frag(aRow1 + k);
        #pragma unroll
        for (int j = 0; j < 4; j++) {
            if (!nOk[j]) continue;
            v16bf b = load_frag(bRow[j] + k);
            acc[0][j] = wmma_bf16(a0, b, acc[0][j]);
            acc[1][j] = wmma_bf16(a1, b, acc[1][j]);
        }
    }

    #pragma unroll
    for (int i = 0; i < 2; i++) {
        #pragma unroll
        for (int j = 0; j < 4; j++) {
            if (!nOk[j]) continue;
            int r0 = mBase + i * 16 + lh * 8;
            int c0 = nBase + j * 16 + l16;
            #pragma unroll
            for (int r = 0; r < 8; r++)
                store_elem(&C[(long)(r0 + r) * ldc + c0], acc[i][j][r]);
        }
    }
}

// ---------------- elementwise / conversion kernels ----------------
__global__ void f32_to_bf16_kernel(const float* __restrict__ in,
                                   unsigned short* __restrict__ out, long n) {
    long i = (long)blockIdx.x * blockDim.x + threadIdx.x;
    long stride = (long)gridDim.x * blockDim.x;
    for (; i < n; i += stride) out[i] = f2bf(in[i]);
}

// wkv_b [H,256,512] f32 -> wkT [H,512,128] bf16 (transposed K part) + wv [H,128,512] bf16
__global__ void split_wkvb_kernel(const float* __restrict__ wkv_b,
                                  unsigned short* __restrict__ wkT,
                                  unsigned short* __restrict__ wv) {
    long n = (long)H_ * 256 * 512;
    long i = (long)blockIdx.x * blockDim.x + threadIdx.x;
    long stride = (long)gridDim.x * blockDim.x;
    for (; i < n; i += stride) {
        long h = i / (256 * 512);
        long r = (i / 512) & 255;
        long c = i & 511;
        unsigned short v = f2bf(wkv_b[i]);
        if (r < 128) wkT[(h * 512 + c) * 128 + r] = v;
        else         wv[(h * 128 + (r - 128)) * 512 + c] = v;
    }
}

// In-place RoPE on the q_pe slice of the q buffer [4096, 3072] (bf16).
__global__ void rope_q_kernel(unsigned short* __restrict__ q,
                              const float* __restrict__ cosT,
                              const float* __restrict__ sinT) {
    int m = blockIdx.x;            // token row 0..4095
    int h = blockIdx.y;
    int s = m & (S_ - 1);
    int i = threadIdx.x;           // pair 0..31
    unsigned short* p = q + (long)m * NQ_ + h * QK_HEAD_ + D_NOPE_ + 2 * i;
    float re = bf2f(p[0]), im = bf2f(p[1]);
    float c = cosT[s * 32 + i], sn = sinT[s * 32 + i];
    p[0] = f2bf(re * c - im * sn);
    p[1] = f2bf(re * sn + im * c);
}

// kv_full f32 [4096,576] -> kvn bf16 [4096,512] (tanh norm) + k_pe bf16 [4096,64] (RoPE)
__global__ void kv_post_kernel(const float* __restrict__ kvf,
                               unsigned short* __restrict__ kvn,
                               unsigned short* __restrict__ kpe,
                               const float* __restrict__ cosT,
                               const float* __restrict__ sinT,
                               const float* __restrict__ alpha,
                               const float* __restrict__ gamma,
                               const float* __restrict__ beta) {
    int m = blockIdx.x;
    int s = m & (S_ - 1);
    int t = threadIdx.x;           // 0..63
    const float* row = kvf + (long)m * 576;
    float a = alpha[0];
    #pragma unroll
    for (int j = 0; j < 8; j++) {
        int c = t * 8 + j;
        float v = tanhf(a * row[c]) * gamma[c] + beta[c];
        kvn[(long)m * 512 + c] = f2bf(v);
    }
    if (t < 32) {
        float re = row[512 + 2 * t], im = row[512 + 2 * t + 1];
        float c = cosT[s * 32 + t], sn = sinT[s * 32 + t];
        kpe[(long)m * 64 + 2 * t]     = f2bf(re * c - im * sn);
        kpe[(long)m * 64 + 2 * t + 1] = f2bf(re * sn + im * c);
    }
}

// One-time transpose: kvn [B][S][512] -> kvnT [B][512][S] (bf16), 32x32 LDS tiles.
__global__ __launch_bounds__(256) void transpose_kvn_kernel(
    const unsigned short* __restrict__ kvn,
    unsigned short* __restrict__ kvnT) {
    __shared__ unsigned short tile[32][33];
    int bz = blockIdx.z;
    int s0 = blockIdx.x * 32;
    int c0 = blockIdx.y * 32;
    int j  = threadIdx.x & 31;
    int i0 = threadIdx.x >> 5;     // 0..7
    const unsigned short* src = kvn + ((long)bz * S_ + s0) * 512 + c0;
    #pragma unroll
    for (int i = i0; i < 32; i += 8)
        tile[i][j] = src[(long)i * 512 + j];
    __syncthreads();
    unsigned short* dst = kvnT + ((long)bz * 512 + c0) * S_ + s0;
    #pragma unroll
    for (int i = i0; i < 32; i += 8)
        dst[(long)i * S_ + j] = tile[j][i];
}

// ---------------- attention: two-pass streaming softmax, WMMA everywhere ----------------
#define P_PAD 72    // probs LDS row stride (shorts): 144B, 16B aligned, bank-friendly

__global__ __launch_bounds__(256) void mla_attn_kernel(
    const unsigned short* __restrict__ qbuf,   // [4096, 3072] (nope+roped pe per head)
    const unsigned short* __restrict__ qabs,   // [H][4096][512]
    const unsigned short* __restrict__ kvn,    // [4096][512]
    const unsigned short* __restrict__ kvnT,   // [B][512][2048]
    const unsigned short* __restrict__ kpe,    // [4096][64]
    unsigned short* __restrict__ olat)         // [H][4096][512]
{
    __shared__ __attribute__((aligned(16))) float partM[2][4][16];
    __shared__ __attribute__((aligned(16))) float partL[2][4][16];
    __shared__ __attribute__((aligned(16))) float gM[32];
    __shared__ __attribute__((aligned(16))) float gInvL[32];
    __shared__ __attribute__((aligned(16))) unsigned short probsS[2][32][P_PAD];

    int h       = blockIdx.y;
    int mBaseWG = blockIdx.x * 32;     // 32 queries per workgroup
    int bIdx    = mBaseWG >> 11;       // batch
    int tGlob0  = bIdx << 11;          // key row base for this batch

    int wave = threadIdx.x >> 5;
    int lane = threadIdx.x & 31;
    int lh   = lane >> 4;
    int l16  = lane & 15;
    int koff = lh * 8;
    const float scale = 0.07216878364870322f;  // 1/sqrt(192)

    const unsigned short* qa_h = qabs + (long)h * M_TOT * 512;
    const unsigned short* qpe0 = qbuf + (long)h * QK_HEAD_ + D_NOPE_;

    // 16x16 score tile (queries mBase.., keys tRow..), scaled.
    auto score16 = [&](int mBase, int tRow) -> v8f {
        v8f c = (v8f){0.f,0.f,0.f,0.f,0.f,0.f,0.f,0.f};
        const unsigned short* aRow = qa_h + (long)(mBase + l16) * 512 + koff;
        const unsigned short* bRow = kvn  + (long)(tRow  + l16) * 512 + koff;
        #pragma unroll
        for (int k = 0; k < 512; k += 32) {
            v16bf a  = load_frag(aRow + k);
            v16bf bb = load_frag(bRow + k);
            c = wmma_bf16(a, bb, c);
        }
        const unsigned short* aP = qpe0 + (long)(mBase + l16) * NQ_ + koff;
        const unsigned short* bP = kpe  + (long)(tRow  + l16) * 64  + koff;
        #pragma unroll
        for (int k = 0; k < 64; k += 32) {
            v16bf a  = load_frag(aP + k);
            v16bf bb = load_frag(bP + k);
            c = wmma_bf16(a, bb, c);
        }
        #pragma unroll
        for (int r = 0; r < 8; r++) c[r] *= scale;
        return c;
    };

    // ---- pass 1: row max / row sum over all 2048 keys ----
    {
        int rt1 = wave >> 2;           // waves 0-3 -> rows 0..15, waves 4-7 -> 16..31
        int wg  = wave & 3;
        int mBase = mBaseWG + rt1 * 16;
        float mrow[8], lrow[8];
        #pragma unroll
        for (int r = 0; r < 8; r++) { mrow[r] = -3.0e38f; lrow[r] = 0.f; }

        for (int kt = wg; kt < 128; kt += 4) {      // 16-key subtiles
            v8f sc = score16(mBase, tGlob0 + kt * 16);
            #pragma unroll
            for (int r = 0; r < 8; r++) {
                float v = sc[r];
                float tmax = v;
                #pragma unroll
                for (int off = 8; off >= 1; off >>= 1)
                    tmax = fmaxf(tmax, __shfl_xor(tmax, off, 32));
                float mn = fmaxf(mrow[r], tmax);
                float p = __expf(v - mn);
                float ps = p;
                #pragma unroll
                for (int off = 8; off >= 1; off >>= 1)
                    ps += __shfl_xor(ps, off, 32);
                lrow[r] = lrow[r] * __expf(mrow[r] - mn) + ps;
                mrow[r] = mn;
            }
        }
        if (l16 == 0) {
            #pragma unroll
            for (int r = 0; r < 8; r++) {
                partM[rt1][wg][lh * 8 + r] = mrow[r];
                partL[rt1][wg][lh * 8 + r] = lrow[r];
            }
        }
    }
    __syncthreads();
    if (threadIdx.x < 32) {
        int rt1 = threadIdx.x >> 4, rr = threadIdx.x & 15;
        float Mv = -3.0e38f;
        #pragma unroll
        for (int wg = 0; wg < 4; wg++) Mv = fmaxf(Mv, partM[rt1][wg][rr]);
        float Lv = 0.f;
        #pragma unroll
        for (int wg = 0; wg < 4; wg++)
            Lv += partL[rt1][wg][rr] * __expf(partM[rt1][wg][rr] - Mv);
        gM[threadIdx.x]    = Mv;
        gInvL[threadIdx.x] = (Lv > 0.f) ? (1.f / Lv) : 0.f;
    }
    __syncthreads();

    // ---- pass 2: o_lat = exp(S - M) @ kvn over 64-key tiles ----
    // Every wave both stages one 16x16 probs subtile and accumulates a
    // 16(row) x 128(channel) slice of o_lat; kvnT gives contiguous B frags.
    int rt2    = wave & 1;             // query row tile (also used for staging)
    int sth    = wave >> 1;            // staging key subtile 0..3
    int cslice = (wave >> 1) * 128;    // channel slice for accumulation
    v8f oacc[8];
    #pragma unroll
    for (int j = 0; j < 8; j++) oacc[j] = (v8f){0.f,0.f,0.f,0.f,0.f,0.f,0.f,0.f};

    const unsigned short* kvT_b = kvnT + (long)bIdx * 512 * S_;   // [512][2048]

    for (int kt = 0; kt < 32; kt++) {                // 64 keys per tile
        unsigned short (*pb)[P_PAD] = probsS[kt & 1];

        // stage: all 8 waves compute one probs subtile each (18 WMMA apiece)
        {
            v8f sc = score16(mBaseWG + rt2 * 16, tGlob0 + kt * 64 + sth * 16);
            #pragma unroll
            for (int r = 0; r < 8; r++) {
                int rowL = rt2 * 16 + lh * 8 + r;
                float p = __expf(sc[r] - gM[rowL]);
                pb[rowL][sth * 16 + l16] = f2bf(p);
            }
        }
        __syncthreads();

        // accumulate: K = 64 keys, B frags straight from global kvnT (L2 resident)
        #pragma unroll
        for (int ct = 0; ct < 8; ct++) {
            int cc = cslice + ct * 16;
            const unsigned short* bRow =
                kvT_b + (long)(cc + l16) * S_ + kt * 64 + koff;
            #pragma unroll
            for (int kk = 0; kk < 64; kk += 32) {
                v16bf a  = load_frag(&pb[rt2 * 16 + l16][kk + koff]);
                v16bf bb = load_frag(bRow + kk);
                oacc[ct] = wmma_bf16(a, bb, oacc[ct]);
            }
        }
        // no second barrier: double-buffered probsS; next iteration's stage
        // writes the other buffer, and its barrier orders reuse of this one.
    }

    // epilogue: divide by row sum, store bf16 o_lat
    #pragma unroll
    for (int ct = 0; ct < 8; ct++) {
        int cc = cslice + ct * 16 + l16;
        #pragma unroll
        for (int r = 0; r < 8; r++) {
            int rowL = rt2 * 16 + lh * 8 + r;
            float v = oacc[ct][r] * gInvL[rowL];
            olat[((long)h * M_TOT + mBaseWG + rowL) * 512 + cc] = f2bf(v);
        }
    }
}

// ---------------- orchestration ----------------
extern "C" void kernel_launch(void* const* d_in, const int* in_sizes, int n_in,
                              void* d_out, int out_size, void* d_ws, size_t ws_size,
                              hipStream_t stream) {
    const float* x      = (const float*)d_in[0];
    const float* cosT   = (const float*)d_in[1];
    const float* sinT   = (const float*)d_in[2];
    const float* wq     = (const float*)d_in[3];
    const float* wkv_a  = (const float*)d_in[4];
    const float* wkv_b  = (const float*)d_in[5];
    const float* wo     = (const float*)d_in[6];
    const float* alpha  = (const float*)d_in[7];
    const float* gamma  = (const float*)d_in[8];
    const float* beta   = (const float*)d_in[9];
    float* out = (float*)d_out;

    char* ws = (char*)d_ws;
    auto alloc = [&](size_t bytes) -> void* {
        void* p = (void*)ws;
        ws += (bytes + 255) & ~(size_t)255;
        return p;
    };
    unsigned short* x_bf    = (unsigned short*)alloc((size_t)M_TOT * DIM_ * 2);
    unsigned short* wq_bf   = (unsigned short*)alloc((size_t)NQ_ * DIM_ * 2);
    unsigned short* wkva_bf = (unsigned short*)alloc((size_t)(KV_RANK_ + D_ROPE_) * DIM_ * 2);
    unsigned short* wo_bf   = (unsigned short*)alloc((size_t)DIM_ * DIM_ * 2);
    unsigned short* wkT_bf  = (unsigned short*)alloc((size_t)H_ * 512 * 128 * 2);
    unsigned short* wv_bf   = (unsigned short*)alloc((size_t)H_ * 128 * 512 * 2);
    unsigned short* q_bf    = (unsigned short*)alloc((size_t)M_TOT * NQ_ * 2);
    float*          kvf     = (float*)alloc((size_t)M_TOT * 576 * 4);
    unsigned short* kvn_bf  = (unsigned short*)alloc((size_t)M_TOT * 512 * 2);
    unsigned short* kvnT_bf = (unsigned short*)alloc((size_t)B_ * 512 * S_ * 2);
    unsigned short* kpe_bf  = (unsigned short*)alloc((size_t)M_TOT * 64 * 2);
    unsigned short* qabs_bf = (unsigned short*)alloc((size_t)H_ * M_TOT * 512 * 2);
    unsigned short* olat_bf = (unsigned short*)alloc((size_t)H_ * M_TOT * 512 * 2);
    unsigned short* o_bf    = (unsigned short*)alloc((size_t)M_TOT * DIM_ * 2);

    // 1) convert inputs/weights to bf16
    f32_to_bf16_kernel<<<2048, 256, 0, stream>>>(x,     x_bf,    (long)M_TOT * DIM_);
    f32_to_bf16_kernel<<<2048, 256, 0, stream>>>(wq,    wq_bf,   (long)NQ_ * DIM_);
    f32_to_bf16_kernel<<<1024, 256, 0, stream>>>(wkv_a, wkva_bf, (long)(KV_RANK_ + D_ROPE_) * DIM_);
    f32_to_bf16_kernel<<<2048, 256, 0, stream>>>(wo,    wo_bf,   (long)DIM_ * DIM_);
    split_wkvb_kernel<<<2048, 256, 0, stream>>>(wkv_b, wkT_bf, wv_bf);

    // 2) q = x @ wq^T  (4096 x 3072, K=2048) -> bf16
    gemm_wmma_kernel<unsigned short><<<dim3(12, 64, 1), 256, 0, stream>>>(
        x_bf, DIM_, 0, wq_bf, DIM_, 0, q_bf, NQ_, 0, M_TOT, NQ_, DIM_);
    // 3) RoPE on q_pe (in place)
    rope_q_kernel<<<dim3(M_TOT, H_), 32, 0, stream>>>(q_bf, cosT, sinT);

    // 4) kv_full = x @ wkv_a^T (4096 x 576) -> f32
    gemm_wmma_kernel<float><<<dim3(3, 64, 1), 256, 0, stream>>>(
        x_bf, DIM_, 0, wkva_bf, DIM_, 0, kvf, 576, 0, M_TOT, 576, DIM_);
    // 5) tanh-norm + RoPE(k_pe)
    kv_post_kernel<<<M_TOT, 64, 0, stream>>>(kvf, kvn_bf, kpe_bf, cosT, sinT,
                                             alpha, gamma, beta);
    // 5b) one-time global transpose of kvn for the probs@kvn WMMA B side
    transpose_kvn_kernel<<<dim3(S_ / 32, 512 / 32, B_), 256, 0, stream>>>(
        kvn_bf, kvnT_bf);

    // 6) q_abs[h] = q_nope[h] @ wkT[h]^T  (4096 x 512, K=128), batched over heads
    gemm_wmma_kernel<unsigned short><<<dim3(2, 64, H_), 256, 0, stream>>>(
        q_bf, NQ_, QK_HEAD_,                 // A: head offset h*192 within q row
        wkT_bf, 128, (long)512 * 128,
        qabs_bf, 512, (long)M_TOT * 512,
        M_TOT, 512, 128);

    // 7) attention -> o_lat bf16 [H][4096][512]
    mla_attn_kernel<<<dim3(M_TOT / 32, H_), 256, 0, stream>>>(
        q_bf, qabs_bf, kvn_bf, kvnT_bf, kpe_bf, olat_bf);

    // 8) o[h] = o_lat[h] @ wv[h]^T  (4096 x 128, K=512) -> o_bf columns h*128..
    gemm_wmma_kernel<unsigned short><<<dim3(1, 64, H_), 256, 0, stream>>>(
        olat_bf, 512, (long)M_TOT * 512,
        wv_bf, 512, (long)128 * 512,
        o_bf, DIM_, 128,                      // C batch stride = 128 cols per head
        M_TOT, 128, 512);

    // 9) out = o @ wo^T  (4096 x 2048, K=2048) -> f32
    gemm_wmma_kernel<float><<<dim3(8, 64, 1), 256, 0, stream>>>(
        o_bf, DIM_, 0, wo_bf, DIM_, 0, out, DIM_, 0, M_TOT, DIM_, DIM_);

    (void)in_sizes; (void)n_in; (void)out_size; (void)ws_size;
}
